// SGCN_59150289600760
// MI455X (gfx1250) — matrix operand order repaired and checked by reading, not compiled, not run on hardware
//
#include <hip/hip_runtime.h>
#include <hip/hip_bf16.h>

// Problem constants
#define NB 128
#define CB 256
#define TT 64
#define VV 32
#define FF 256
#define TB 4                 // t-values per workgroup
#define NTB (TT / TB)        // 16 t-blocks
#define COLS (TB * VV)       // 128 h-columns per block (t,v flattened)
#define TV (TT * VV)         // 2048, c-row stride in x

typedef __attribute__((ext_vector_type(16))) __bf16 v16bf;
typedef __attribute__((ext_vector_type(8)))  __bf16 v8bf;
typedef __attribute__((ext_vector_type(8)))  float  v8f;
typedef __attribute__((ext_vector_type(4)))  float  v4f;
typedef __attribute__((ext_vector_type(4)))  int    v4i;

// LDS layout (bytes, all strides multiples of 16B for ds_load_b128)
#define XT_STRIDE 528                       // 256 bf16 (c) + 16B pad
#define XT_BYTES  (COLS * XT_STRIDE)        // 67584 : x transposed [col][c]
#define H_STRIDE  272                       // 128 bf16 (cols) + 16B pad
#define H_BYTES   (FF * H_STRIDE)           // 69632 : h [f][col]
#define AT_BYTES  (TB * VV * VV * 2)        // 8192  : a transposed [t][w][v]
#define ARAW_BYTES (TB * VV * VV * 4)       // 16384 : raw fp32 a tile (async staging)
// 161792 bytes total -> still 2 blocks per 320KB WGP
#define SMEM_BYTES (XT_BYTES + H_BYTES + AT_BYTES + ARAW_BYTES)

// Async global<->LDS path (ASYNCcnt). Guarded: falls back to VGPR copy if the
// builtins are absent in this toolchain. (__has_builtin is false on the host
// pass too, but SMEM_BYTES is identical either way.)
#if defined(__HIP_DEVICE_COMPILE__) && \
    __has_builtin(__builtin_amdgcn_global_load_async_to_lds_b128) && \
    __has_builtin(__builtin_amdgcn_global_store_async_from_lds_b128)
#define USE_ASYNC_LDS 1
#else
#define USE_ASYNC_LDS 0
#endif

// b128 async ops take int4 pointers: global side is AS(1), LDS side is AS(3)
typedef __attribute__((address_space(1))) v4i GV4;
typedef __attribute__((address_space(3))) v4i LV4;

__device__ __forceinline__ unsigned pack2(float lo, float hi) {
  __bf16 l = (__bf16)lo, h = (__bf16)hi;
  unsigned short ls, hs;
  __builtin_memcpy(&ls, &l, 2);
  __builtin_memcpy(&hs, &h, 2);
  return ((unsigned)hs << 16) | (unsigned)ls;
}

// Wt[f][c] = bf16(W[c][f]) : tiny, stays L2-resident, reused by all blocks
__global__ void wt_kernel(const float* __restrict__ W, __bf16* __restrict__ Wt) {
  int f = blockIdx.x, c = threadIdx.x;
  Wt[f * FF + c] = (__bf16)W[(size_t)c * FF + f];
}

__global__ __launch_bounds__(256)
void sgcn_fused(const float* __restrict__ x, const float* __restrict__ a,
                const float* __restrict__ bias, const __bf16* __restrict__ Wt,
                float* __restrict__ out, float* __restrict__ outA) {
  extern __shared__ char smem[];
  char* xT   = smem;                                   // [col][c]  bf16
  char* hL   = smem + XT_BYTES;                        // [f][col]  bf16
  char* aT   = smem + XT_BYTES + H_BYTES;              // [t][w][v] bf16
  char* aRAW = smem + XT_BYTES + H_BYTES + AT_BYTES;   // [t][v][w] fp32

  const int tid = threadIdx.x;
  const int wg  = blockIdx.x;
  const int n   = wg / NTB;
  const int tb  = wg % NTB;
  const int t0  = tb * TB;

  const float* xbase  = x    + ((size_t)n * CB * TT + (size_t)t0) * VV; // +c*TV+col
  const float* abase  = a    + ((size_t)n * TT + t0) * (VV * VV);
  float*       oAbase = outA + ((size_t)n * TT + t0) * (VV * VV);

  // hint the streaming x slice into cache (global_prefetch_b8)
  __builtin_prefetch(xbase + (size_t)tid * TV, 0, 0);

#if USE_ASYNC_LDS
  // ---- async fetch of the fp32 a-tile straight into LDS (ASYNCcnt path)
#pragma unroll
  for (int i = 0; i < 4; ++i) {
    int la = (i * 256 + tid) * 4;
    __builtin_amdgcn_global_load_async_to_lds_b128(
        (GV4*)(abase + la), (LV4*)(aRAW + (size_t)la * 4), 0, 0);
  }
#endif

  // ---- stage x-slice: coalesced fp32 loads, bf16 c-pair packed, transposed to xT[col][c]
#pragma unroll
  for (int i = 0; i < 16; ++i) {
    int u    = i * 256 + tid;                      // 0..4095
    int c2   = u >> 5;                             // c-pair 0..127
    int col4 = (u & 31) * 4;                       // 0..124
    const float* g0 = xbase + (size_t)(2 * c2) * TV + col4;
    v4f r0 = *(const v4f*)g0;
    v4f r1 = *(const v4f*)(g0 + TV);
#pragma unroll
    for (int k = 0; k < 4; ++k)
      *(unsigned*)(xT + (col4 + k) * XT_STRIDE + c2 * 4) = pack2(r0[k], r1[k]);
  }

  // ---- a-tile: fp32 copy to output #2 + bf16 transpose into LDS aT[t][w][v]
#if USE_ASYNC_LDS
#if __has_builtin(__builtin_amdgcn_s_wait_asynccnt)
  __builtin_amdgcn_s_wait_asynccnt(0);   // this wave's async loads are in LDS
#else
  asm volatile("s_wait_asynccnt 0x0" ::: "memory");
#endif
  __syncthreads();                        // make every wave's tile visible
#pragma unroll
  for (int i = 0; i < 4; ++i) {
    int la = (i * 256 + tid) * 4;
    v4f r = *(const v4f*)(aRAW + (size_t)la * 4);      // ds_load_b128
    // fp32 passthrough copy: LDS -> global, no VGPR round-trip
    __builtin_amdgcn_global_store_async_from_lds_b128(
        (GV4*)(oAbase + la), (LV4*)(aRAW + (size_t)la * 4), 0, 0);
    int t = la >> 10, v = (la >> 5) & 31, w = la & 31;
    char* p = aT + t * 2048 + w * 64 + v * 2;
#pragma unroll
    for (int k = 0; k < 4; ++k)
      *(__bf16*)(p + k * 64) = (__bf16)r[k];           // (w+k) rows
  }
#else
#pragma unroll
  for (int i = 0; i < 4; ++i) {
    int la = (i * 256 + tid) * 4;
    v4f r = *(const v4f*)(abase + la);
    *(v4f*)(oAbase + la) = r;
    int t = la >> 10, v = (la >> 5) & 31, w = la & 31;
    char* p = aT + t * 2048 + w * 64 + v * 2;
#pragma unroll
    for (int k = 0; k < 4; ++k)
      *(__bf16*)(p + k * 64) = (__bf16)r[k];
  }
#endif
  __syncthreads();

  const int wave = tid >> 5;
  const int lane = tid & 31;
  const int l15  = lane & 15;
  const int hi   = lane >> 4;          // lane half select
  const int fbase = wave * 32;         // each wave owns 32 f-rows (2 M-tiles)

  // ================= step 1: h = Wt(bf16) x xT(bf16), f32 accum =================
  v8f acc[2][8];
#pragma unroll
  for (int mi = 0; mi < 2; ++mi)
#pragma unroll
    for (int j = 0; j < 8; ++j)
      acc[mi][j] = (v8f)(0.0f);

  for (int kk = 0; kk < 8; ++kk) {     // K = C in chunks of 32
    v16bf afrag[2];
#pragma unroll
    for (int mi = 0; mi < 2; ++mi) {
      // A layout: lanes0-15 K{0..7,16..23}, lanes16-31 K{8..15,24..31}
      const __bf16* wp = Wt + (size_t)(fbase + mi * 16 + l15) * 256 + kk * 32 + hi * 8;
      v8bf alo = *(const v8bf*)wp;
      v8bf ahi = *(const v8bf*)(wp + 16);
#pragma unroll
      for (int q = 0; q < 8; ++q) { afrag[mi][q] = alo[q]; afrag[mi][q + 8] = ahi[q]; }
    }
#pragma unroll
    for (int j = 0; j < 8; ++j) {
      // B layout: lanes0-15 N=l15,K{0..7,16..23} ; lanes16-31 N=l15,K{8..15,24..31}
      const char* bp = xT + (j * 16 + l15) * XT_STRIDE + kk * 64 + hi * 16;
      v8bf blo = *(const v8bf*)bp;
      v8bf bhi = *(const v8bf*)(bp + 32);
      v16bf bfrag;
#pragma unroll
      for (int q = 0; q < 8; ++q) { bfrag[q] = blo[q]; bfrag[q + 8] = bhi[q]; }
#pragma unroll
      for (int mi = 0; mi < 2; ++mi)
        acc[mi][j] = __builtin_amdgcn_wmma_f32_16x16x32_bf16(
            false, afrag[mi], false, bfrag, (short)0, acc[mi][j], false, false);
    }
  }

  // ---- add bias, h -> bf16 into hL (each wave's private f-rows; no barrier needed)
#pragma unroll
  for (int mi = 0; mi < 2; ++mi) {
#pragma unroll
    for (int r = 0; r < 8; ++r) {
      int f = fbase + mi * 16 + r + hi * 8;        // C/D layout: M = r (+8 for hi lanes)
      float bv = bias[f];
      char* hp = hL + f * H_STRIDE + l15 * 2;      // N = l15
#pragma unroll
      for (int j = 0; j < 8; ++j)
        *(__bf16*)(hp + j * 32) = (__bf16)(acc[mi][j][r] + bv);
    }
  }

  // ================= step 2: out = h(bf16) x aT(bf16), f32 accum =================
#pragma unroll
  for (int t = 0; t < TB; ++t) {
    v16bf hfrag[2];
#pragma unroll
    for (int mi = 0; mi < 2; ++mi) {
      const char* hp = hL + (fbase + mi * 16 + l15) * H_STRIDE + t * 64 + hi * 16;
      v8bf lo  = *(const v8bf*)hp;
      v8bf hi8 = *(const v8bf*)(hp + 32);
#pragma unroll
      for (int q = 0; q < 8; ++q) { hfrag[mi][q] = lo[q]; hfrag[mi][q + 8] = hi8[q]; }
    }
#pragma unroll
    for (int wt = 0; wt < 2; ++wt) {
      const char* bp = aT + t * 2048 + (wt * 16 + l15) * 64 + hi * 16;
      v8bf blo = *(const v8bf*)bp;
      v8bf bhi = *(const v8bf*)(bp + 32);
      v16bf bfrag;
#pragma unroll
      for (int q = 0; q < 8; ++q) { bfrag[q] = blo[q]; bfrag[q + 8] = bhi[q]; }
#pragma unroll
      for (int mi = 0; mi < 2; ++mi) {
        v8f o = (v8f)(0.0f);
        o = __builtin_amdgcn_wmma_f32_16x16x32_bf16(
            false, hfrag[mi], false, bfrag, (short)0, o, false, false);
#pragma unroll
        for (int r = 0; r < 8; ++r) {
          int f = fbase + mi * 16 + r + hi * 8;
          out[(((size_t)n * FF + f) * TT + (t0 + t)) * VV + wt * 16 + l15] = o[r];
        }
      }
    }
  }
  // outstanding async stores (if any) drain under S_ENDPGM's implicit wait-idle
}

extern "C" void kernel_launch(void* const* d_in, const int* in_sizes, int n_in,
                              void* d_out, int out_size, void* d_ws, size_t ws_size,
                              hipStream_t stream) {
  const float* x = (const float*)d_in[0];   // [128,256,64,32]
  const float* a = (const float*)d_in[1];   // [128,64,32,32]
  const float* W = (const float*)d_in[2];   // [256,256]
  const float* b = (const float*)d_in[3];   // [256]
  float* out  = (float*)d_out;                              // out1: 128*256*64*32
  float* outA = out + (size_t)NB * FF * TT * VV;            // out2: a copy
  __bf16* Wt = (__bf16*)d_ws;               // 128 KB scratch for bf16(W^T)

  wt_kernel<<<dim3(FF), dim3(CB), 0, stream>>>(W, Wt);
  sgcn_fused<<<dim3(NB * NTB), dim3(256), SMEM_BYTES, stream>>>(x, a, b, Wt, out, outA);
}
